// BiLSTM_CRF_14800457302253
// MI455X (gfx1250) — matrix operand order repaired and checked by reading, not compiled, not run on hardware
//
#include <hip/hip_runtime.h>
#include <hip/hip_bf16.h>
#include <math.h>

#define EMB   1024
#define H2    512
#define NGATE 2048          // 4*H2, PyTorch gate order i,f,g,o
#define NTAG  5
#define START_TAG 3
#define STOP_TAG  4
#define NEGV (-10000.0f)
#define NWG 64              // workgroups per direction for the recurrent scan
#define UNITS_PER_WG (H2 / NWG)          // 8 hidden units per WG
#define ROWS_PER_WG  (4 * UNITS_PER_WG)  // 32 gate rows per WG

typedef unsigned short u16;
typedef __attribute__((ext_vector_type(4)))  unsigned short u16x4;
typedef __attribute__((ext_vector_type(8)))  unsigned short u16x8;
typedef __attribute__((ext_vector_type(16))) __bf16        v16bf;
typedef __attribute__((ext_vector_type(8)))  float         v8f;

union V16 { u16x8 h[2]; v16bf v; };

__device__ __forceinline__ u16 f2bf(float f) {
  unsigned u = __float_as_uint(f);
  u += 0x7FFFu + ((u >> 16) & 1u);          // round-to-nearest-even
  return (u16)(u >> 16);
}
__device__ __forceinline__ float bf2f(u16 h) {
  return __uint_as_float(((unsigned)h) << 16);
}
__device__ __forceinline__ float sigm(float x) { return 1.0f / (1.0f + __expf(-x)); }

// ---------------------------------------------------------------------------
// Init: zero the two per-direction step counters (d_ws is poisoned by harness)
// ---------------------------------------------------------------------------
__global__ void k_init(unsigned* ctr) {
  if (threadIdx.x < 2) ctr[threadIdx.x] = 0u;
}

// ---------------------------------------------------------------------------
// Embedding gather + f32 -> bf16 convert: xbf[t,:] = bf16(emb[sentence[t],:])
// ---------------------------------------------------------------------------
__global__ void k_embed_gather_bf16(const int* __restrict__ sent,
                                    const float* __restrict__ emb,
                                    u16* __restrict__ xbf) {
  int t = blockIdx.x;
  int row = sent[t];
  const float4* src = (const float4*)(emb + (size_t)row * EMB);
  u16x4* dst = (u16x4*)(xbf + (size_t)t * EMB);
  for (int i = threadIdx.x; i < EMB / 4; i += blockDim.x) {
    float4 v = src[i];
    u16x4 o;
    o[0] = f2bf(v.x); o[1] = f2bf(v.y); o[2] = f2bf(v.z); o[3] = f2bf(v.w);
    dst[i] = o;
  }
}

// Generic f32 -> bf16 weight convert
__global__ void k_f32_to_bf16(const float* __restrict__ in,
                              u16* __restrict__ out, int n) {
  int i = blockIdx.x * blockDim.x + threadIdx.x;
  int stride = gridDim.x * blockDim.x;
  for (; i < n; i += stride) out[i] = f2bf(in[i]);
}

// ---------------------------------------------------------------------------
// Batched input GEMM via WMMA:  G[dir][t, r] = sum_k x[t,k]*W_ih[r,k] + b[r]
// A = x tile (16 x 32 bf16), B = W_ih rows (K-major, 32 x 16), C/D f32.
// Fragment layouts per CDNA5 ISA 7.12.2 (wave32).
// grid = (T/16, NGATE/128, 2), block = 256 (8 waves, one 16x16 tile each)
// ---------------------------------------------------------------------------
__global__ __launch_bounds__(256)
void k_wmma_gemm(const u16* __restrict__ xbf,
                 const u16* __restrict__ wihF, const u16* __restrict__ wihB,
                 const float* __restrict__ bF, const float* __restrict__ bB,
                 float* __restrict__ gF, float* __restrict__ gB) {
  const int wave = threadIdx.x >> 5;
  const int lane = threadIdx.x & 31;
  const int m    = lane & 15;       // A row (time) / B col (gate) / D col
  const int half = lane >> 4;       // lane-half selects K sub-blocks
  const int t0   = blockIdx.x * 16;
  const int n0   = (blockIdx.y * 8 + wave) * 16;
  const int dir  = blockIdx.z;

  const u16*   W    = dir ? wihB : wihF;
  const float* bias = dir ? bB   : bF;
  float*       G    = dir ? gB   : gF;

  const u16* Arow = xbf + (size_t)(t0 + m) * EMB;
  const u16* Brow = W   + (size_t)(n0 + m) * EMB;

  v8f acc = {};
  for (int kb = 0; kb < EMB; kb += 32) {
    V16 a, b;
    // A 16x32 bf16: lanes 0-15 K={kb..kb+7, kb+16..kb+23}, lanes 16-31 +8
    a.h[0] = *(const u16x8*)(Arow + kb + half * 8);
    a.h[1] = *(const u16x8*)(Arow + kb + 16 + half * 8);
    // B 32x16 bf16 (K-major in memory): lanes 0-15 K={kb..kb+7, kb+16..kb+23}, lanes 16-31 +8
    b.h[0] = *(const u16x8*)(Brow + kb + half * 8);
    b.h[1] = *(const u16x8*)(Brow + kb + 16 + half * 8);
    __builtin_prefetch(Arow + kb + 512, 0, 0);   // global_prefetch_b8
    __builtin_prefetch(Brow + kb + 512, 0, 0);
    acc = __builtin_amdgcn_wmma_f32_16x16x32_bf16(
        /*neg_a=*/false, a.v, /*neg_b=*/false, b.v,
        /*c_mod=*/(short)0, acc, /*reuse_a=*/false, /*reuse_b=*/false);
  }

  // D 16x16 f32: row M = vgpr + 8*half, col N = lane&15
  float bv = bias[n0 + m];
#pragma unroll
  for (int r = 0; r < 8; ++r) {
    int row = t0 + half * 8 + r;
    G[(size_t)row * NGATE + n0 + m] = acc[r] + bv;
  }
}

// ---------------------------------------------------------------------------
// Persistent recurrent scan. 2*NWG workgroups; each owns UNITS_PER_WG hidden
// units (4 gate rows each) with its W_hh slice resident in LDS (32 KB bf16).
// Per step: read 2KB h vector, LDS matvec, activations, publish h, device-
// scope counter barrier (atomics + s_sleep spin).
// ---------------------------------------------------------------------------
__global__ __launch_bounds__(256)
void k_lstm_scan(const u16* __restrict__ whhF, const u16* __restrict__ whhB,
                 const float* __restrict__ gF, const float* __restrict__ gB,
                 const float* __restrict__ h0, const float* __restrict__ c0,
                 float* __restrict__ Hf, float* __restrict__ HbRev,
                 float* __restrict__ hbuf, unsigned* __restrict__ ctr, int T) {
  const int dir = blockIdx.x / NWG;
  const int wg  = blockIdx.x % NWG;
  const int tid = threadIdx.x;
  const int j0  = wg * UNITS_PER_WG;

  const u16*   Whh = dir ? whhB : whhF;
  const float* G   = dir ? gB : gF;
  float*       Hs  = dir ? HbRev : Hf;
  float*       hvec = hbuf + dir * H2;
  unsigned*    cnt  = ctr + dir;

  __shared__ u16   Wl[ROWS_PER_WG * H2];   // 32 KB bf16 W_hh slice
  __shared__ float hsh[H2];
  __shared__ float gsh[ROWS_PER_WG];

  // Preload W_hh slice (rows: gate gi, unit u -> global row gi*H2 + j0 + u)
  for (int e = tid; e < ROWS_PER_WG * H2 / 8; e += blockDim.x) {
    int rr = e / (H2 / 8);
    int c8 = e % (H2 / 8);
    int gi = rr >> 3, u = rr & 7;
    int R  = gi * H2 + j0 + u;
    *(u16x8*)(&Wl[rr * H2 + c8 * 8]) =
        *(const u16x8*)(Whh + (size_t)R * H2 + c8 * 8);
  }
  float cu = 0.0f;
  if (tid < UNITS_PER_WG) cu = c0[dir * H2 + j0 + tid];
  __syncthreads();

  const int rr  = tid >> 3;   // gate row 0..31 (8 lanes cooperate per row)
  const int sub = tid & 7;    // 64-column strip per lane
  const int gi  = rr >> 3;
  const int u   = rr & 7;

  for (int s = 0; s < T; ++s) {
    const float* hprev = (s == 0) ? (h0 + dir * H2) : hvec;
    for (int j = tid; j < H2; j += blockDim.x)
      hsh[j] = ((volatile const float*)hprev)[j];
    __syncthreads();

    float sum = 0.0f;
    const int cbase = sub * 64;
#pragma unroll 2
    for (int cc = 0; cc < 64; cc += 8) {
      u16x8 wv = *(const u16x8*)(&Wl[rr * H2 + cbase + cc]);  // ds_load_b128
#pragma unroll
      for (int q = 0; q < 8; ++q)
        sum += bf2f(wv[q]) * hsh[cbase + cc + q];
    }
    sum += __shfl_xor(sum, 1);
    sum += __shfl_xor(sum, 2);
    sum += __shfl_xor(sum, 4);

    int tg = dir ? (T - 1 - s) : s;         // backward scans reversed x
    if (sub == 0)
      gsh[rr] = G[(size_t)tg * NGATE + gi * H2 + j0 + u] + sum;
    __syncthreads();

    if (tid < UNITS_PER_WG) {
      float iv = sigm(gsh[tid]);
      float fv = sigm(gsh[8 + tid]);
      float gv = tanhf(gsh[16 + tid]);
      float ov = sigm(gsh[24 + tid]);
      cu = fv * cu + iv * gv;
      float hv = ov * tanhf(cu);
      hvec[j0 + tid] = hv;
      Hs[(size_t)s * H2 + j0 + tid] = hv;
    }
    __threadfence();
    __syncthreads();
    if (tid == 0) {
      __hip_atomic_fetch_add(cnt, 1u, __ATOMIC_ACQ_REL, __HIP_MEMORY_SCOPE_AGENT);
      unsigned target = (unsigned)(s + 1) * NWG;
      while (__hip_atomic_load(cnt, __ATOMIC_ACQUIRE,
                               __HIP_MEMORY_SCOPE_AGENT) < target)
        __builtin_amdgcn_s_sleep(1);
    }
    __syncthreads();
  }
}

// ---------------------------------------------------------------------------
// feats[t,k] = b_out[k] + hf[t]·W_out[k,:512] + hb[t]·W_out[k,512:]
// hb[t] = HbRev[T-1-t]
// ---------------------------------------------------------------------------
__global__ __launch_bounds__(256)
void k_proj(const float* __restrict__ Hf, const float* __restrict__ HbRev,
            const float* __restrict__ Wout, const float* __restrict__ bout,
            float* __restrict__ feats, int T) {
  int t = blockIdx.x;
  int tid = threadIdx.x;
  __shared__ float red[NTAG][256];
  float p[NTAG] = {};
  for (int j = tid; j < H2; j += 256) {
    float a = Hf[(size_t)t * H2 + j];
    float b = HbRev[(size_t)(T - 1 - t) * H2 + j];
#pragma unroll
    for (int k = 0; k < NTAG; ++k)
      p[k] += a * Wout[k * (2 * H2) + j] + b * Wout[k * (2 * H2) + H2 + j];
  }
  for (int k = 0; k < NTAG; ++k) red[k][tid] = p[k];
  __syncthreads();
  for (int off = 128; off > 0; off >>= 1) {
    if (tid < off)
      for (int k = 0; k < NTAG; ++k) red[k][tid] += red[k][tid + off];
    __syncthreads();
  }
  if (tid < NTAG) feats[(size_t)t * NTAG + tid] = red[tid][0] + bout[tid];
}

// ---------------------------------------------------------------------------
// Viterbi: one wave32; lanes 0..4 carry per-tag scores, shuffles broadcast.
// out[0]=score, out[1..T]=path (as float, per harness output dtype).
// ---------------------------------------------------------------------------
__global__ void k_viterbi(const float* __restrict__ feats,
                          const float* __restrict__ trans,
                          int* __restrict__ bptr,
                          float* __restrict__ out, int T, int out_size) {
  int lane = threadIdx.x;
  float trow[NTAG] = {};
  if (lane < NTAG)
    for (int f = 0; f < NTAG; ++f) trow[f] = trans[lane * NTAG + f];

  float fv = (lane == START_TAG) ? 0.0f : NEGV;
  for (int t = 0; t < T; ++t) {
    float best = -3.4e38f; int bi = 0;
#pragma unroll
    for (int f = 0; f < NTAG; ++f) {
      float v = __shfl(fv, f) + ((lane < NTAG) ? trow[f] : 0.0f);
      if (v > best) { best = v; bi = f; }      // strict > keeps first argmax
    }
    if (lane < NTAG) {
      bptr[t * NTAG + lane] = bi;
      fv = best + feats[(size_t)t * NTAG + lane];
    }
  }
  float term = fv + ((lane < NTAG) ? trans[STOP_TAG * NTAG + lane] : NEGV);

  float score = -3.4e38f; int btag = 0;
#pragma unroll
  for (int f = 0; f < NTAG; ++f) {            // all lanes active for shuffles
    float v = __shfl(term, f);
    if (v > score) { score = v; btag = f; }
  }
  if (lane == 0) {
    if (out_size > 0) out[0] = score;
    int cur = btag;
    if (T < out_size) out[T] = (float)cur;    // path[T-1]
    for (int t = T - 1; t >= 1; --t) {
      cur = bptr[t * NTAG + cur];
      if (t < out_size) out[t] = (float)cur;  // path[t-1] -> out[1+(t-1)]
    }
  }
}

// ---------------------------------------------------------------------------
extern "C" void kernel_launch(void* const* d_in, const int* in_sizes, int n_in,
                              void* d_out, int out_size, void* d_ws, size_t ws_size,
                              hipStream_t stream) {
  const int*   sent  = (const int*)  d_in[0];
  const float* emb   = (const float*)d_in[1];
  const float* WihF  = (const float*)d_in[2];
  const float* WhhF  = (const float*)d_in[3];
  const float* bF    = (const float*)d_in[4];
  const float* WihB  = (const float*)d_in[5];
  const float* WhhB  = (const float*)d_in[6];
  const float* bB    = (const float*)d_in[7];
  const float* Wout  = (const float*)d_in[8];
  const float* bout  = (const float*)d_in[9];
  const float* trans = (const float*)d_in[10];
  const float* h0    = (const float*)d_in[11];
  const float* c0    = (const float*)d_in[12];
  const int T = in_sizes[0];
  (void)n_in; (void)ws_size;

  char* ws = (char*)d_ws;
  size_t off = 0;
  auto alloc = [&](size_t bytes) -> char* {
    char* p = ws + off;
    off = (off + bytes + 255) & ~(size_t)255;
    return p;
  };
  unsigned* ctr   = (unsigned*)alloc(256);
  u16*   xbf      = (u16*)  alloc((size_t)T * EMB * 2);
  u16*   wihFb    = (u16*)  alloc((size_t)NGATE * EMB * 2);
  u16*   wihBb    = (u16*)  alloc((size_t)NGATE * EMB * 2);
  u16*   whhFb    = (u16*)  alloc((size_t)NGATE * H2 * 2);
  u16*   whhBb    = (u16*)  alloc((size_t)NGATE * H2 * 2);
  float* gF       = (float*)alloc((size_t)T * NGATE * 4);
  float* gB       = (float*)alloc((size_t)T * NGATE * 4);
  float* Hf       = (float*)alloc((size_t)T * H2 * 4);
  float* HbRev    = (float*)alloc((size_t)T * H2 * 4);
  float* feats    = (float*)alloc((size_t)T * NTAG * 4);
  int*   bptr     = (int*)  alloc((size_t)T * NTAG * 4);
  float* hbuf     = (float*)alloc(2 * H2 * 4);

  hipLaunchKernelGGL(k_init, dim3(1), dim3(32), 0, stream, ctr);
  hipLaunchKernelGGL(k_embed_gather_bf16, dim3(T), dim3(256), 0, stream,
                     sent, emb, xbf);
  hipLaunchKernelGGL(k_f32_to_bf16, dim3(1024), dim3(256), 0, stream,
                     WihF, wihFb, NGATE * EMB);
  hipLaunchKernelGGL(k_f32_to_bf16, dim3(1024), dim3(256), 0, stream,
                     WihB, wihBb, NGATE * EMB);
  hipLaunchKernelGGL(k_f32_to_bf16, dim3(512), dim3(256), 0, stream,
                     WhhF, whhFb, NGATE * H2);
  hipLaunchKernelGGL(k_f32_to_bf16, dim3(512), dim3(256), 0, stream,
                     WhhB, whhBb, NGATE * H2);

  // WMMA batched input GEMM (both directions), bias folded in
  hipLaunchKernelGGL(k_wmma_gemm, dim3(T / 16, NGATE / 128, 2), dim3(256), 0,
                     stream, xbf, wihFb, wihBb, bF, bB, gF, gB);

  // Persistent device-synced recurrent scan (both directions concurrently)
  hipLaunchKernelGGL(k_lstm_scan, dim3(2 * NWG), dim3(256), 0, stream,
                     whhFb, whhBb, gF, gB, h0, c0, Hf, HbRev, hbuf, ctr, T);

  hipLaunchKernelGGL(k_proj, dim3(T), dim3(256), 0, stream,
                     Hf, HbRev, Wout, bout, feats, T);

  hipLaunchKernelGGL(k_viterbi, dim3(1), dim3(32), 0, stream,
                     feats, trans, bptr, (float*)d_out, T, out_size);
}
